// Block_69664369541369
// MI455X (gfx1250) — matrix operand order repaired and checked by reading, not compiled, not run on hardware
//
#include <hip/hip_runtime.h>
#include <hip/hip_bf16.h>

typedef __attribute__((ext_vector_type(16))) _Float16 v16h;
typedef __attribute__((ext_vector_type(8)))  _Float16 v8h;
typedef __attribute__((ext_vector_type(8)))  float    v8f;

#define BB 8
#define TT 1024
#define CC 1024
#define HH 16
#define HS 64
#define FF 4096

__device__ __forceinline__ v8f wmma16(v16h a, v16h b, v8f c) {
    return __builtin_amdgcn_wmma_f32_16x16x32_f16(false, a, false, b, (short)0, c, false, false);
}

__device__ __forceinline__ v8f vzero8() {
    v8f r;
#pragma unroll
    for (int i = 0; i < 8; ++i) r[i] = 0.0f;
    return r;
}

// ---- CDNA5 async global->LDS copy (ASYNCcnt-tracked, bypasses VGPRs) ----
// VDST VGPR holds the LDS byte address (flat->LDS mapping = addr[31:0]).
__device__ __forceinline__ void cp16_async(_Float16* dst_lds, const _Float16* src_glb) {
    unsigned lds = (unsigned)(size_t)dst_lds;
    asm volatile("global_load_async_to_lds_b128 %0, %1, off"
                 :: "v"(lds), "v"(src_glb)
                 : "memory");
}

template<int N>
__device__ __forceinline__ void wait_async() {
    asm volatile("s_wait_asynccnt %0" :: "i"(N) : "memory");
}

// ---- WMMA fragment loaders (ISA 16-bit layouts, wave32) ----
// A 16x32 (MxK): lane l (row = l&15, sel = l>>4) holds K = sel*8 + j (j<8)
// and K = 16 + sel*8 + (j-8) (j>=8): two contiguous 8-half chunks.
__device__ __forceinline__ v16h load_frag_A(const _Float16* __restrict__ base, int ld) {
    int lane = threadIdx.x & 31;
    int row = lane & 15, sel = lane >> 4;
    const _Float16* p = base + (size_t)row * ld;
    v8h a0 = *(const v8h*)(p + sel * 8);
    v8h a1 = *(const v8h*)(p + 16 + sel * 8);
    v16h r;
#pragma unroll
    for (int j = 0; j < 8; ++j) { r[j] = a0[j]; r[8 + j] = a1[j]; }
    return r;
}

// B 32x16 (KxN): lane l (n = l&15, sel = l>>4) holds K = sel*8 + j (j<8)
// and K = 16 + sel*8 + (j-8) (j>=8). With source stored row-major [N][K]
// (i.e. W[out][in]), these are two contiguous 8-half chunks of row n.
__device__ __forceinline__ v16h load_frag_B(const _Float16* __restrict__ base, int ld) {
    int lane = threadIdx.x & 31;
    int n = lane & 15, sel = lane >> 4;
    const _Float16* p = base + (size_t)n * ld;
    v8h b0 = *(const v8h*)(p + sel * 8);
    v8h b1 = *(const v8h*)(p + 16 + sel * 8);
    v16h r;
#pragma unroll
    for (int j = 0; j < 8; ++j) { r[j] = b0[j]; r[8 + j] = b1[j]; }
    return r;
}

// ---- weight fake-quant -> f16 ----
__global__ __launch_bounds__(256) void quant_w_kernel(
    const float* __restrict__ W, const float* __restrict__ e, const float* __restrict__ b,
    _Float16* __restrict__ out, long n, int K)
{
    long i = (long)blockIdx.x * 256 + threadIdx.x;
    if (i >= n) return;
    long row = i / K;
    float ee = e[row];
    float br = fmaxf(b[row], 0.0f);
    float mn = (br > 0.0f) ? -exp2f(br - 1.0f) : 0.0f;
    float mx = (br > 0.0f) ?  exp2f(br - 1.0f) - 1.0f : 0.0f;
    float qw = fminf(fmaxf(exp2f(-ee) * W[i], mn), mx);
    out[i] = (_Float16)(exp2f(ee) * rintf(qw));
}

// ---- layernorm: f32 [rows][C] -> f16 ----
__global__ __launch_bounds__(256) void ln_kernel(
    const float* __restrict__ x, const float* __restrict__ g, const float* __restrict__ be,
    _Float16* __restrict__ out)
{
    int row = blockIdx.x;
    const float* xr = x + (size_t)row * CC;
    int t = threadIdx.x;
    float v[4];
    float s = 0.0f;
#pragma unroll
    for (int i = 0; i < 4; ++i) { v[i] = xr[t + i * 256]; s += v[i]; }
    __shared__ float red[8];
#pragma unroll
    for (int o = 16; o > 0; o >>= 1) s += __shfl_xor(s, o, 32);
    if ((t & 31) == 0) red[t >> 5] = s;
    __syncthreads();
    float tot = 0.0f;
#pragma unroll
    for (int i = 0; i < 8; ++i) tot += red[i];
    float mu = tot * (1.0f / CC);
    float sq = 0.0f;
#pragma unroll
    for (int i = 0; i < 4; ++i) { float d = v[i] - mu; sq += d * d; }
#pragma unroll
    for (int o = 16; o > 0; o >>= 1) sq += __shfl_xor(sq, o, 32);
    __syncthreads();
    if ((t & 31) == 0) red[t >> 5] = sq;
    __syncthreads();
    float tot2 = 0.0f;
#pragma unroll
    for (int i = 0; i < 8; ++i) tot2 += red[i];
    float rstd = rsqrtf(tot2 * (1.0f / CC) + 1e-5f);
    _Float16* o16 = out + (size_t)row * CC;
#pragma unroll
    for (int i = 0; i < 4; ++i) {
        int c = t + i * 256;
        o16[c] = (_Float16)((v[i] - mu) * rstd * g[c] + be[c]);
    }
}

// ---- tiled WMMA GEMM: out[M][N] = act[M][K] @ W[N][K]^T, epilogue by MODE ----
// Double-buffered LDS, async global->LDS pipeline (ASYNCcnt).
// MODE 0: store f16. MODE 1: relu(acc+bias)->f16. MODE 2: acc+bias+resid->f32.
#define LDT 40
template<int MODE>
__global__ __launch_bounds__(256) void gemm_kernel(
    const _Float16* __restrict__ A, const _Float16* __restrict__ W,
    int M, int N, int K,
    _Float16* __restrict__ out16, float* __restrict__ out32,
    const float* __restrict__ bias, const float* __restrict__ resid)
{
    __shared__ __align__(16) _Float16 As[2][128 * LDT];
    __shared__ __align__(16) _Float16 Bs[2][128 * LDT];
    int t = threadIdx.x;
    int lane = t & 31, wave = t >> 5;
    int wm = wave >> 2, wn = wave & 3;          // 2x4 waves -> wave tile 64x32
    const _Float16* Ag = A + (size_t)blockIdx.x * 128 * K;
    const _Float16* Wg = W + (size_t)blockIdx.y * 128 * K;
    int lrow = t >> 1, lcol = (t & 1) * 16;

    v8f acc[4][2];
#pragma unroll
    for (int mt = 0; mt < 4; ++mt)
#pragma unroll
        for (int nt = 0; nt < 2; ++nt) acc[mt][nt] = vzero8();

    auto issue = [&](int ki, int buf) {
        const _Float16* ap = Ag + (size_t)lrow * K + ki * 32 + lcol;
        const _Float16* bp = Wg + (size_t)lrow * K + ki * 32 + lcol;
        cp16_async(&As[buf][lrow * LDT + lcol],     ap);
        cp16_async(&As[buf][lrow * LDT + lcol + 8], ap + 8);
        cp16_async(&Bs[buf][lrow * LDT + lcol],     bp);
        cp16_async(&Bs[buf][lrow * LDT + lcol + 8], bp + 8);
        __builtin_prefetch(ap + 32, 0, 0);   // L2 prefetch one more tile ahead
        __builtin_prefetch(bp + 32, 0, 0);
    };

    const int nk = K / 32;
    issue(0, 0);
    for (int ki = 0; ki < nk; ++ki) {
        int buf = ki & 1;
        if (ki + 1 < nk) {
            issue(ki + 1, buf ^ 1);
            wait_async<4>();                 // tile ki landed; ki+1 in flight
        } else {
            wait_async<0>();
        }
        __syncthreads();                     // all waves see tile ki in LDS
        v16h af[4], bf[2];
#pragma unroll
        for (int mt = 0; mt < 4; ++mt)
            af[mt] = load_frag_A(&As[buf][(wm * 64 + mt * 16) * LDT], LDT);
#pragma unroll
        for (int nt = 0; nt < 2; ++nt)
            bf[nt] = load_frag_B(&Bs[buf][(wn * 32 + nt * 16) * LDT], LDT);
#pragma unroll
        for (int mt = 0; mt < 4; ++mt)
#pragma unroll
            for (int nt = 0; nt < 2; ++nt)
                acc[mt][nt] = wmma16(af[mt], bf[nt], acc[mt][nt]);
        __syncthreads();                     // done reading buf before it is reused
    }

    int sel = lane >> 4, nn = lane & 15;
#pragma unroll
    for (int mt = 0; mt < 4; ++mt) {
#pragma unroll
        for (int nt = 0; nt < 2; ++nt) {
            int col = blockIdx.y * 128 + wn * 32 + nt * 16 + nn;
#pragma unroll
            for (int r = 0; r < 8; ++r) {
                int row = blockIdx.x * 128 + wm * 64 + mt * 16 + sel * 8 + r;
                float v = acc[mt][nt][r];
                if (MODE == 0) {
                    out16[(size_t)row * N + col] = (_Float16)v;
                } else if (MODE == 1) {
                    out16[(size_t)row * N + col] = (_Float16)fmaxf(v + bias[col], 0.0f);
                } else {
                    out32[(size_t)row * N + col] = v + bias[col] + resid[(size_t)row * N + col];
                }
            }
        }
    }
}

// ---- transpose v [B,T,H*HS] -> vt [B,H,HS,T] ----
__global__ __launch_bounds__(256) void transpose_v_kernel(
    const _Float16* __restrict__ v, _Float16* __restrict__ vt)
{
    size_t i = (size_t)blockIdx.x * 256 + threadIdx.x;
    int s = (int)(i & (TT - 1));
    size_t r = i >> 10;
    int d = (int)(r & (HS - 1));
    size_t r2 = r >> 6;
    int h = (int)(r2 & (HH - 1));
    int b = (int)(r2 >> 4);
    vt[i] = v[((size_t)b * TT + s) * CC + h * HS + d];
}

// ---- flash attention (causal), one wave = 16 q rows ----
__global__ __launch_bounds__(256) void attn_kernel(
    const _Float16* __restrict__ q, const _Float16* __restrict__ k,
    const _Float16* __restrict__ vt, _Float16* __restrict__ att)
{
    int bh = blockIdx.x;
    int b = bh >> 4, h = bh & 15;
    int wave = threadIdx.x >> 5, lane = threadIdx.x & 31;
    int q0 = blockIdx.y * 128 + wave * 16;
    int sel = lane >> 4, nn = lane & 15;

    __shared__ __align__(16) _Float16 Pst[8 * 16 * LDT];
    _Float16* Pw = &Pst[wave * 16 * LDT];

    const _Float16* qbase = q + ((size_t)b * TT + q0) * CC + h * HS;
    v16h qf0 = load_frag_A(qbase, CC);
    v16h qf1 = load_frag_A(qbase + 32, CC);

    float m_i[8], l_i[8];
    v8f oacc[4];
#pragma unroll
    for (int r = 0; r < 8; ++r) { m_i[r] = -1e30f; l_i[r] = 0.0f; }
#pragma unroll
    for (int dt = 0; dt < 4; ++dt) oacc[dt] = vzero8();

    const _Float16* vtb = vt + ((size_t)bh * HS) * TT;

    for (int s0 = 0; s0 < q0 + 16; s0 += 32) {
        float sv[2][8];
#pragma unroll
        for (int st = 0; st < 2; ++st) {
            int scol = s0 + st * 16;
            v8f sc = vzero8();
            if (scol <= q0 + 15) {   // wave-uniform predicate
                const _Float16* kb = k + ((size_t)b * TT + scol) * CC + h * HS;
                v16h bf0 = load_frag_B(kb, CC);
                v16h bf1 = load_frag_B(kb + 32, CC);
                sc = wmma16(qf0, bf0, sc);
                sc = wmma16(qf1, bf1, sc);
            }
#pragma unroll
            for (int r = 0; r < 8; ++r) {
                int trow = q0 + sel * 8 + r;
                int ss = scol + nn;
                sv[st][r] = (scol <= q0 + 15 && ss <= trow) ? sc[r] * 0.03125f : -1e30f;
            }
        }
        // online softmax: row max / sum across the 16-lane group
#pragma unroll
        for (int r = 0; r < 8; ++r) {
            float mt = fmaxf(sv[0][r], sv[1][r]);
#pragma unroll
            for (int msk = 1; msk < 16; msk <<= 1) mt = fmaxf(mt, __shfl_xor(mt, msk, 32));
            float mnew = fmaxf(m_i[r], mt);
            float e0 = __expf(sv[0][r] - mnew);
            float e1 = __expf(sv[1][r] - mnew);
            float ls = e0 + e1;
#pragma unroll
            for (int msk = 1; msk < 16; msk <<= 1) ls += __shfl_xor(ls, msk, 32);
            float corr = __expf(m_i[r] - mnew);
            l_i[r] = l_i[r] * corr + ls;
            m_i[r] = mnew;
#pragma unroll
            for (int dt = 0; dt < 4; ++dt) oacc[dt][r] *= corr;
            int m = sel * 8 + r;
            Pw[m * LDT + nn]      = (_Float16)e0;
            Pw[m * LDT + 16 + nn] = (_Float16)e1;
        }
        v16h pf = load_frag_A(Pw, LDT);
#pragma unroll
        for (int dt = 0; dt < 4; ++dt) {
            v16h vf = load_frag_B(vtb + (size_t)(dt * 16) * TT + s0, TT);
            oacc[dt] = wmma16(pf, vf, oacc[dt]);
        }
    }

#pragma unroll
    for (int dt = 0; dt < 4; ++dt) {
#pragma unroll
        for (int r = 0; r < 8; ++r) {
            int trow = q0 + sel * 8 + r;
            att[((size_t)b * TT + trow) * CC + h * HS + dt * 16 + nn] =
                (_Float16)(oacc[dt][r] / l_i[r]);
        }
    }
}

extern "C" void kernel_launch(void* const* d_in, const int* in_sizes, int n_in,
                              void* d_out, int out_size, void* d_ws, size_t ws_size,
                              hipStream_t stream) {
    const float* x     = (const float*)d_in[0];
    const float* Wq    = (const float*)d_in[1];
    const float* Wk    = (const float*)d_in[2];
    const float* Wv    = (const float*)d_in[3];
    const float* eq    = (const float*)d_in[4];
    const float* bq    = (const float*)d_in[5];
    const float* ek    = (const float*)d_in[6];
    const float* bk    = (const float*)d_in[7];
    const float* ev    = (const float*)d_in[8];
    const float* bv    = (const float*)d_in[9];
    const float* Wp    = (const float*)d_in[10];
    const float* ep    = (const float*)d_in[11];
    const float* bp    = (const float*)d_in[12];
    const float* biasp = (const float*)d_in[13];
    const float* W1    = (const float*)d_in[14];
    const float* e1    = (const float*)d_in[15];
    const float* b1    = (const float*)d_in[16];
    const float* bias1 = (const float*)d_in[17];
    const float* W2    = (const float*)d_in[18];
    const float* e2    = (const float*)d_in[19];
    const float* b2    = (const float*)d_in[20];
    const float* bias2 = (const float*)d_in[21];
    const float* g1    = (const float*)d_in[22];
    const float* be1   = (const float*)d_in[23];
    const float* g2    = (const float*)d_in[24];
    const float* be2   = (const float*)d_in[25];

    char* ws = (char*)d_ws;
    size_t off = 0;
    auto alloc = [&](size_t bytes) -> void* {
        off = (off + 255) & ~(size_t)255;
        void* p = ws + off;
        off += bytes;
        return p;
    };
    const size_t MT = (size_t)BB * TT;          // 8192 rows
    _Float16* wq16 = (_Float16*)alloc((size_t)CC * CC * 2);
    _Float16* wk16 = (_Float16*)alloc((size_t)CC * CC * 2);
    _Float16* wv16 = (_Float16*)alloc((size_t)CC * CC * 2);
    _Float16* wp16 = (_Float16*)alloc((size_t)CC * CC * 2);
    _Float16* w116 = (_Float16*)alloc((size_t)FF * CC * 2);
    _Float16* w216 = (_Float16*)alloc((size_t)CC * FF * 2);
    _Float16* h16  = (_Float16*)alloc(MT * CC * 2);
    _Float16* q16  = (_Float16*)alloc(MT * CC * 2);
    _Float16* k16  = (_Float16*)alloc(MT * CC * 2);
    _Float16* v16  = (_Float16*)alloc(MT * CC * 2);
    _Float16* vt16 = (_Float16*)alloc(MT * CC * 2);
    _Float16* att16= (_Float16*)alloc(MT * CC * 2);
    float*    x2   = (float*)   alloc(MT * CC * 4);
    _Float16* h2   = h16;                        // alias: h16 dead after QKV
    _Float16* ff16 = q16;                        // alias: q/k/v/vt (64MB) dead after attention

    // 1) quantize weights to f16
    long nqkv = (long)CC * CC;
    quant_w_kernel<<<(int)((nqkv + 255) / 256), 256, 0, stream>>>(Wq, eq, bq, wq16, nqkv, CC);
    quant_w_kernel<<<(int)((nqkv + 255) / 256), 256, 0, stream>>>(Wk, ek, bk, wk16, nqkv, CC);
    quant_w_kernel<<<(int)((nqkv + 255) / 256), 256, 0, stream>>>(Wv, ev, bv, wv16, nqkv, CC);
    quant_w_kernel<<<(int)((nqkv + 255) / 256), 256, 0, stream>>>(Wp, ep, bp, wp16, nqkv, CC);
    long n1 = (long)FF * CC;
    quant_w_kernel<<<(int)((n1 + 255) / 256), 256, 0, stream>>>(W1, e1, b1, w116, n1, CC);
    quant_w_kernel<<<(int)((n1 + 255) / 256), 256, 0, stream>>>(W2, e2, b2, w216, n1, FF);

    // 2) LN1
    ln_kernel<<<(int)MT, 256, 0, stream>>>(x, g1, be1, h16);

    // 3) QKV projections
    dim3 g_1024((int)MT / 128, CC / 128);
    gemm_kernel<0><<<g_1024, 256, 0, stream>>>(h16, wq16, (int)MT, CC, CC, q16, nullptr, nullptr, nullptr);
    gemm_kernel<0><<<g_1024, 256, 0, stream>>>(h16, wk16, (int)MT, CC, CC, k16, nullptr, nullptr, nullptr);
    gemm_kernel<0><<<g_1024, 256, 0, stream>>>(h16, wv16, (int)MT, CC, CC, v16, nullptr, nullptr, nullptr);

    // 4) transpose V -> [B,H,HS,T]
    transpose_v_kernel<<<(int)(MT * CC / 256), 256, 0, stream>>>(v16, vt16);

    // 5) causal flash attention
    attn_kernel<<<dim3(BB * HH, TT / 128), 256, 0, stream>>>(q16, k16, vt16, att16);

    // 6) output projection + bias + residual -> x2 (f32)
    gemm_kernel<2><<<g_1024, 256, 0, stream>>>(att16, wp16, (int)MT, CC, CC, nullptr, x2, biasp, x);

    // 7) LN2
    ln_kernel<<<(int)MT, 256, 0, stream>>>(x2, g2, be2, h2);

    // 8) FFN1: relu(h2 @ W1^T + bias1) -> f16
    dim3 g_ff((int)MT / 128, FF / 128);
    gemm_kernel<1><<<g_ff, 256, 0, stream>>>(h2, w116, (int)MT, FF, CC, ff16, nullptr, bias1, nullptr);

    // 9) FFN2: x2 + ff @ W2^T + bias2 -> d_out (f32)
    gemm_kernel<2><<<g_1024, 256, 0, stream>>>(ff16, w216, (int)MT, CC, FF, nullptr, (float*)d_out, bias2, x2);
}
